// CrossAttentionFusion_8143257993861
// MI455X (gfx1250) — matrix operand order repaired and checked by reading, not compiled, not run on hardware
//
#include <hip/hip_runtime.h>
#include <hip/hip_bf16.h>

typedef _Float16 v16h __attribute__((ext_vector_type(16)));
typedef _Float16 v8h  __attribute__((ext_vector_type(8)));
typedef _Float16 v4h  __attribute__((ext_vector_type(4)));
typedef float    v8f  __attribute__((ext_vector_type(8)));

#define C_       256
#define HW_      16384      // N = H*W
#define CHUNK_   2048
#define BLKM     64         // query rows per workgroup
#define KB       32         // keys per iteration
#define THREADS  128        // 4 waves (wave32)
#define KSTRIDE  264        // Klds row stride (halves): C_+8 pad -> conflict-free
#define VSTRIDE  40         // Vt  row stride (halves): KB+8 pad
#define PSTRIDE  40         // P   row stride (halves)
#define SCALE_   0.0625f    // 1/sqrt(256)

// LDS layout (bytes); epilogue aliases the front of this pool per-wave.
#define KLDS_OFF   0
#define KLDS_BYTES (KB * KSTRIDE * 2)            // 16896
#define VLDS_OFF   (KLDS_OFF + KLDS_BYTES)
#define VLDS_BYTES (C_ * VSTRIDE * 2)            // 20480
#define PLDS_OFF   (VLDS_OFF + VLDS_BYTES)       // 37376 (16B aligned)
#define PLDS_BYTES (4 * 16 * PSTRIDE * 2)        // 5120
#define SMEM_BYTES (PLDS_OFF + PLDS_BYTES)       // 42496

static __device__ __forceinline__ v16h cat8(v8h lo, v8h hi) {
  return __builtin_shufflevector(lo, hi, 0,1,2,3,4,5,6,7,8,9,10,11,12,13,14,15);
}
static __device__ __forceinline__ v8f wmma16(v16h a, v16h b, v8f c) {
  // v_wmma_f32_16x16x32_f16
  return __builtin_amdgcn_wmma_f32_16x16x32_f16(false, a, false, b, (short)0, c,
                                                false, false);
}

__global__ __launch_bounds__(THREADS, 1)
void cross_attn_fused_kernel(const float* __restrict__ lidar,
                             const float* __restrict__ image,
                             const float* __restrict__ mw,
                             float* __restrict__ out) {
  __shared__ __align__(16) char smem[SMEM_BYTES];
  _Float16* Klds = (_Float16*)(smem + KLDS_OFF);   // 32 keys x 256 chan, row-major
  _Float16* Vt   = (_Float16*)(smem + VLDS_OFF);   // 256 chan x 32 keys (transposed)
  _Float16* Plds = (_Float16*)(smem + PLDS_OFF);   // per-wave 16x32 P bounce

  const int tid   = threadIdx.x;
  const int lane  = tid & 31;
  const int wid   = tid >> 5;
  const int lrow  = lane & 15;
  const int hi    = lane >> 4;                     // 0 or 1 (half-wave)
  const int chunk = blockIdx.y;
  const int b     = blockIdx.z;

  const int n0    = chunk * CHUNK_;
  const int bbase = b * C_ * HW_;
  const int qrow0 = blockIdx.x * BLKM + wid * 16;  // wave's first query row in chunk
  const int qglob = n0 + qrow0 + lrow;             // this lane's query position in N

  // ---- Q fragments: A-layout (16x32 f16). Data is channel-major, so these are
  // strided f32 gathers done ONCE, then reused for 64 key iterations from VGPRs.
  v16h qfrag[8];
  {
    const float* qb = image + bbase + qglob;       // + c*HW_ walks channels
    const int hs = hi * 8;                         // K offset for this half-wave
#pragma unroll
    for (int f = 0; f < 8; ++f) {
      const int cb = f * 32;
      v16h a;
#pragma unroll
      for (int j = 0; j < 8; ++j) a[j]     = (_Float16)qb[(cb + hs + j) * HW_];
#pragma unroll
      for (int j = 0; j < 8; ++j) a[8 + j] = (_Float16)qb[(cb + 16 + hs + j) * HW_];
      qfrag[f] = a;
    }
  }

  // ---- flash-attention running state (rows live in 16-lane groups: M = r + 8*hi)
  v8f Oacc[16];
#pragma unroll
  for (int t = 0; t < 16; ++t) Oacc[t] = {};
  float mrow[8], lsum[8];
#pragma unroll
  for (int r = 0; r < 8; ++r) { mrow[r] = -1e30f; lsum[r] = 0.0f; }

  const int woff = wid * 16 * PSTRIDE;

  for (int kb = 0; kb < CHUNK_; kb += KB) {
    __syncthreads();
    // ---- stage K block (f32 -> f16) into row-major Klds (for S=QK^T) and
    // transposed Vt (for P@V). One float4 = 4 keys of one channel; 128B global
    // line = one channel's whole 32-key run (fully coalesced). V == K.
    {
      const float4* src4 = (const float4*)(lidar + bbase + n0 + kb);
#pragma unroll
      for (int i = 0; i < (C_ * KB / 4) / THREADS; ++i) {   // 16 iters
        const int idx = i * THREADS + tid;                  // 0..2047
        const int c   = idx >> 3;
        const int k   = (idx & 7) * 4;
        const float4 v = src4[c * (HW_ / 4) + (idx & 7)];
        const _Float16 h0 = (_Float16)v.x, h1 = (_Float16)v.y;
        const _Float16 h2 = (_Float16)v.z, h3 = (_Float16)v.w;
        Klds[(k + 0) * KSTRIDE + c] = h0;
        Klds[(k + 1) * KSTRIDE + c] = h1;
        Klds[(k + 2) * KSTRIDE + c] = h2;
        Klds[(k + 3) * KSTRIDE + c] = h3;
        *(v4h*)(Vt + c * VSTRIDE + k) = (v4h){h0, h1, h2, h3};  // ds_store_b64
      }
    }
    __syncthreads();

    // ---- prefetch next key block into cache while WMMAs run (256 lines; each
    // line = one channel row). No counter cost: global_prefetch_b8.
    if (kb + KB < CHUNK_) {
      const float* nsrc = lidar + bbase + n0 + kb + KB;
      __builtin_prefetch(nsrc + tid * HW_, 0, 0);
      __builtin_prefetch(nsrc + (tid + THREADS) * HW_, 0, 0);
    }

    // ---- scores: two 16x16 tiles (keys kb..+15, kb+16..+31), K-dim = C = 256.
    // B-layout 32x16: lane<16 holds col N=lane, K 0..7 and 16..23 (table order).
    v8f s0 = {}, s1 = {};
    {
      const _Float16* kp0 = Klds + lrow * KSTRIDE + (hi << 3);
      const _Float16* kp1 = kp0 + 16 * KSTRIDE;
#pragma unroll
      for (int cs = 0; cs < C_; cs += 32) {
        v16h b0 = cat8(*(const v8h*)(kp0 + cs), *(const v8h*)(kp0 + cs + 16));
        v16h b1 = cat8(*(const v8h*)(kp1 + cs), *(const v8h*)(kp1 + cs + 16));
        s0 = wmma16(qfrag[cs >> 5], b0, s0);
        s1 = wmma16(qfrag[cs >> 5], b1, s1);
      }
    }

    // ---- online softmax; row reductions = shfl_xor over the 16-lane group
    float cf[8];
#pragma unroll
    for (int r = 0; r < 8; ++r) {
      float a0 = s0[r] * SCALE_;
      float a1 = s1[r] * SCALE_;
      float t = fmaxf(a0, a1);
#pragma unroll
      for (int d = 1; d < 16; d <<= 1) t = fmaxf(t, __shfl_xor(t, d, 32));
      const float mn = fmaxf(mrow[r], t);
      cf[r] = __expf(mrow[r] - mn);
      const float p0 = __expf(a0 - mn);
      const float p1 = __expf(a1 - mn);
      float rs = p0 + p1;
#pragma unroll
      for (int d = 1; d < 16; d <<= 1) rs += __shfl_xor(rs, d, 32);
      lsum[r] = lsum[r] * cf[r] + rs;
      mrow[r] = mn;
      // bounce P to LDS row-major 16x32 (D-layout -> A-layout transpose)
      const int M = r + (hi << 3);
      Plds[woff + M * PSTRIDE + lrow]      = (_Float16)p0;
      Plds[woff + M * PSTRIDE + 16 + lrow] = (_Float16)p1;
    }

    // ---- O = diag(cf)*O + P @ V   (per-wave LDS is in-order: no barrier)
    const _Float16* pp = Plds + woff + lrow * PSTRIDE + hi * 8;
    const v16h pa = cat8(*(const v8h*)pp, *(const v8h*)(pp + 16));
#pragma unroll
    for (int t = 0; t < 16; ++t) {
      const _Float16* vp = Vt + (t * 16 + lrow) * VSTRIDE + (hi << 3);
      const v16h vb = cat8(*(const v8h*)vp, *(const v8h*)(vp + 16));
      v8f o = Oacc[t];
#pragma unroll
      for (int r = 0; r < 8; ++r) o[r] *= cf[r];
      Oacc[t] = wmma16(pa, vb, o);
    }
  }

  // ---- epilogue: normalize, blend with image, coalesce stores via LDS bounce
  __syncthreads();                        // retire K/Vt/P, alias smem per-wave
  float* Olds = (float*)smem + wid * 2048;  // 8KB per wave (fits in pool)
  float inv_l[8];
#pragma unroll
  for (int r = 0; r < 8; ++r) inv_l[r] = 1.0f / lsum[r];

  const float e0 = __expf(mw[0]);
  const float e1 = __expf(mw[1]);
  const float w0 = e0 / (e0 + e1);
  const float w1 = e1 / (e0 + e1);

  const int nq0 = n0 + qrow0;
#pragma unroll
  for (int pass = 0; pass < 2; ++pass) {
#pragma unroll
    for (int t = 0; t < 8; ++t) {
      v8f o = Oacc[pass * 8 + t];
#pragma unroll
      for (int r = 0; r < 8; ++r) {
        const int M = r + (hi << 3);
        Olds[M * 128 + t * 16 + lrow] = o[r] * inv_l[r];
      }
    }
#pragma unroll 4
    for (int it = 0; it < 64; ++it) {     // lanes 0-15 / 16-31: contiguous 64B runs
      const int cl = it * 2 + hi;         // 0..127
      const float val = Olds[lrow * 128 + cl];
      const int g = bbase + (pass * 128 + cl) * HW_ + nq0 + lrow;
      out[g] = w0 * image[g] + w1 * val;
    }
  }
}

extern "C" void kernel_launch(void* const* d_in, const int* in_sizes, int n_in,
                              void* d_out, int out_size, void* d_ws, size_t ws_size,
                              hipStream_t stream) {
  const float* lidar = (const float*)d_in[0];   // keys/values
  const float* image = (const float*)d_in[1];   // queries + residual blend
  const float* mw    = (const float*)d_in[2];   // modality_weights (2)
  float* out = (float*)d_out;
  dim3 grid(CHUNK_ / BLKM, HW_ / CHUNK_, 4);    // (32, 8, 4)
  dim3 block(THREADS);
  cross_attn_fused_kernel<<<grid, block, 0, stream>>>(lidar, image, mw, out);
}
